// Attention_63934883168998
// MI455X (gfx1250) — matrix operand order repaired and checked by reading, not compiled, not run on hardware
//
#include <hip/hip_runtime.h>

// ---------------------------------------------------------------------------
// Per-token "head attention" block on MI455X (gfx1250).
//   q/k/v = x @ W^T  (3x GEMM 16384x1024x1024)        -> v_wmma_f32_16x16x32_bf16
//   per token: 16x16 softmax over heads, O = P @ v    -> f32 VALU (<1% of FLOPs)
//   out = O @ Wo^T   (GEMM 16384x1024x1024)           -> v_wmma_f32_16x16x32_bf16
// Compute-bound: 137 GFLOP of GEMM vs ~6us HBM floor; bf16 WMMA (K=32/instr)
// is ~8x the f32 WMMA (K=4/instr) rate, so inputs are downconverted once.
//
// v3: rounds 1-2 showed the backend capping register pressure at ~128 VGPRs
// (8-waves/SIMD occupancy target), which forced every B fragment into one
// VGPR block and a full `s_wait_loadcnt 0x0` before every WMMA.
//  - amdgpu_waves_per_eu(1) lifts the pressure budget to the 256-VGPR cap
//    (12 B-frags + 12 accumulators + A ~ 215 VGPRs -> still 4 waves/SIMD).
//  - sched_group_barrier pins each K-step to [26x VMEM_READ][12x WMMA] so
//    WMMAs retire on partial load waits while later loads stay in flight.
// ---------------------------------------------------------------------------

typedef __bf16 bf16;
typedef __bf16 v16bf __attribute__((ext_vector_type(16)));
typedef __bf16 v8bf  __attribute__((ext_vector_type(8)));
typedef float  v8f   __attribute__((ext_vector_type(8)));

#define HIDDEN 1024
#define TOKENS 16384   // B(4) * S(4096)

#if defined(__has_builtin)
#if __has_builtin(__builtin_amdgcn_sched_group_barrier)
#define SGB(mask, size, id) __builtin_amdgcn_sched_group_barrier((mask), (size), (id))
#endif
#endif
#ifndef SGB
#define SGB(mask, size, id)
#endif
// sched_group_barrier masks
#define SG_MFMA      0x008   // matrix ops (WMMA on gfx12xx)
#define SG_VMEM_READ 0x020

static __device__ __forceinline__ v8bf ldv8(const bf16* p) {
    return *(const v8bf*)p;   // 16B aligned -> global_load_b128
}
static __device__ __forceinline__ v16bf cat8(v8bf lo, v8bf hi) {
    v16bf r;
#pragma unroll
    for (int i = 0; i < 8; ++i) { r[i] = lo[i]; r[8 + i] = hi[i]; }
    return r;
}

// ---------------------------------------------------------------- f32 -> bf16
__global__ void cvt_f32_bf16(const float* __restrict__ s, bf16* __restrict__ d, int n) {
    int i = blockIdx.x * blockDim.x + threadIdx.x;
    int stride = gridDim.x * blockDim.x;
    for (; i < n; i += stride) d[i] = (bf16)s[i];
}

// ------------------------------------------------------- fused q,k,v GEMM
// Wave job: 16 tokens x 64 cols for q,k AND v (A fragment reused 12x per
// K-step).  A-frag: lane m<16 holds K {e, e+8-swizzle} per ISA 16-bit A layout
// -> two b128 loads.  B-frag: B[k][n] = W[n][k] (y = x W^T), lane holds
// column n = row n of W, same split K order as A -> two b128 loads.
__global__ __launch_bounds__(256)
__attribute__((amdgpu_waves_per_eu(1)))
void qkv_gemm(
    const bf16* __restrict__ xb,
    const bf16* __restrict__ wq, const bf16* __restrict__ wk, const bf16* __restrict__ wv,
    bf16* __restrict__ q, bf16* __restrict__ k, bf16* __restrict__ v)
{
    const int lane    = threadIdx.x & 31;
    const int gid     = blockIdx.x * 8 + (threadIdx.x >> 5); // global wave job
    const int m_tile  = gid >> 4;    // 0..1023  (16 tokens each)
    const int n_group = gid & 15;    // 0..15    (64 cols each)
    const int hi      = lane >> 4;   // lane half selects K sub-block

    const bf16* aptr = xb + (size_t)(m_tile * 16 + (lane & 15)) * HIDDEN + hi * 8;
    const int   ncol = n_group * 64 + (lane & 15);
    const int   bk   = hi * 8;
    // one per-lane base per weight; s-subtile becomes an immediate 32KB offset
    const size_t bbase = (size_t)ncol * HIDDEN + bk;

    v8f accq[4], acck[4], accv[4];
#pragma unroll
    for (int s = 0; s < 4; ++s) {
#pragma unroll
        for (int r = 0; r < 8; ++r) { accq[s][r] = 0.f; acck[s][r] = 0.f; accv[s][r] = 0.f; }
    }

    for (int k0 = 0; k0 < HIDDEN; k0 += 32) {
        // ---- load phase: A + all 12 B fragments, distinct registers ----
        v16bf A = cat8(ldv8(aptr + k0), ldv8(aptr + k0 + 16));
        v16bf Bq[4], Bk[4], Bv[4];
#pragma unroll
        for (int s = 0; s < 4; ++s) {
            const size_t bofs = bbase + (size_t)s * 16 * HIDDEN + k0;
            Bq[s] = cat8(ldv8(wq + bofs), ldv8(wq + bofs + 16));
            Bk[s] = cat8(ldv8(wk + bofs), ldv8(wk + bofs + 16));
            Bv[s] = cat8(ldv8(wv + bofs), ldv8(wv + bofs + 16));
        }
        // ---- compute phase: 12 WMMAs retire on partial load waits ----
#pragma unroll
        for (int s = 0; s < 4; ++s) {
            accq[s] = __builtin_amdgcn_wmma_f32_16x16x32_bf16(false, A, false, Bq[s],
                                                              (short)0, accq[s], false, false);
            acck[s] = __builtin_amdgcn_wmma_f32_16x16x32_bf16(false, A, false, Bk[s],
                                                              (short)0, acck[s], false, false);
            accv[s] = __builtin_amdgcn_wmma_f32_16x16x32_bf16(false, A, false, Bv[s],
                                                              (short)0, accv[s], false, false);
        }
        // pin the schedule: all 26 loads first, then the 12 WMMAs
        SGB(SG_VMEM_READ, 26, 0);
        SGB(SG_MFMA, 12, 0);
    }

    // C layout: VGPR r -> (M = r + 8*hi, N = lane%16)
    const int mrow0 = m_tile * 16 + hi * 8;
#pragma unroll
    for (int s = 0; s < 4; ++s) {
#pragma unroll
        for (int r = 0; r < 8; ++r) {
            const size_t o = (size_t)(mrow0 + r) * HIDDEN + ncol + s * 16;
            q[o] = (bf16)accq[s][r];
            k[o] = (bf16)acck[s][r];
            v[o] = (bf16)accv[s][r];
        }
    }
}

// ---------------------------------------------- per-token 16x16 head-softmax
// 1 wave per token; q/k/v staged to LDS as f32.  Each lane owns one half-row
// of S (8 entries of row h = lane>>1); row reductions via shfl_xor(1).
__global__ __launch_bounds__(128) void attn_heads(
    const bf16* __restrict__ q, const bf16* __restrict__ k,
    const bf16* __restrict__ v, bf16* __restrict__ o)
{
    __shared__ float smem[4][3 * HIDDEN + 256];
    const int lane  = threadIdx.x & 31;
    const int wave  = threadIdx.x >> 5;
    const int token = blockIdx.x * 4 + wave;

    float* sQ = smem[wave];
    float* sK = sQ + HIDDEN;
    float* sV = sK + HIDDEN;
    float* sP = sV + HIDDEN;

    const bf16* qb = q + (size_t)token * HIDDEN;
    const bf16* kb = k + (size_t)token * HIDDEN;
    const bf16* vb = v + (size_t)token * HIDDEN;
    for (int i = lane; i < HIDDEN; i += 32) {
        sQ[i] = (float)qb[i]; sK[i] = (float)kb[i]; sV[i] = (float)vb[i];
    }
    __syncthreads();

    const int h  = lane >> 1;
    const int gb = (lane & 1) * 8;
    float sc[8];
#pragma unroll
    for (int i = 0; i < 8; ++i) {
        const float* qr = sQ + h * 64;
        const float* kr = sK + (gb + i) * 64;
        float d = 0.f;
        for (int t = 0; t < 64; ++t) d = fmaf(qr[t], kr[t], d);
        sc[i] = d * 0.125f;   // 1/sqrt(HEAD_DIM)
    }
    float m = sc[0];
#pragma unroll
    for (int i = 1; i < 8; ++i) m = fmaxf(m, sc[i]);
    m = fmaxf(m, __shfl_xor(m, 1, 32));          // join the two half-rows
    float sum = 0.f;
#pragma unroll
    for (int i = 0; i < 8; ++i) { sc[i] = __expf(sc[i] - m); sum += sc[i]; }
    sum += __shfl_xor(sum, 1, 32);
    const float inv = 1.0f / sum;
#pragma unroll
    for (int i = 0; i < 8; ++i) sP[h * 16 + gb + i] = sc[i] * inv;
    __syncthreads();

    bf16* ob = o + (size_t)token * HIDDEN;
#pragma unroll
    for (int j = 0; j < 32; ++j) {
        const int idx = j * 32 + lane;
        const int hh = idx >> 6, d = idx & 63;
        float acc = 0.f;
#pragma unroll
        for (int g = 0; g < 16; ++g) acc = fmaf(sP[hh * 16 + g], sV[g * 64 + d], acc);
        ob[idx] = (bf16)acc;
    }
}

// ------------------------------------------------------------- out = O @ Wo^T
__global__ __launch_bounds__(256)
__attribute__((amdgpu_waves_per_eu(1)))
void out_gemm(
    const bf16* __restrict__ Ob, const bf16* __restrict__ wo, float* __restrict__ out)
{
    const int lane    = threadIdx.x & 31;
    const int gid     = blockIdx.x * 8 + (threadIdx.x >> 5);
    const int m_tile  = gid >> 4;
    const int n_group = gid & 15;
    const int hi      = lane >> 4;

    const bf16* aptr = Ob + (size_t)(m_tile * 16 + (lane & 15)) * HIDDEN + hi * 8;
    const int   ncol = n_group * 64 + (lane & 15);
    const int   bk   = hi * 8;
    const size_t bbase = (size_t)ncol * HIDDEN + bk;

    v8f acc[4];
#pragma unroll
    for (int s = 0; s < 4; ++s)
#pragma unroll
        for (int r = 0; r < 8; ++r) acc[s][r] = 0.f;

    for (int k0 = 0; k0 < HIDDEN; k0 += 32) {
        v16bf A = cat8(ldv8(aptr + k0), ldv8(aptr + k0 + 16));
        v16bf B[4];
#pragma unroll
        for (int s = 0; s < 4; ++s) {
            const size_t bofs = bbase + (size_t)s * 16 * HIDDEN + k0;
            B[s] = cat8(ldv8(wo + bofs), ldv8(wo + bofs + 16));
        }
#pragma unroll
        for (int s = 0; s < 4; ++s)
            acc[s] = __builtin_amdgcn_wmma_f32_16x16x32_bf16(false, A, false, B[s],
                                                             (short)0, acc[s], false, false);
        SGB(SG_VMEM_READ, 10, 0);
        SGB(SG_MFMA, 4, 0);
    }

    const int mrow0 = m_tile * 16 + hi * 8;
#pragma unroll
    for (int s = 0; s < 4; ++s)
#pragma unroll
        for (int r = 0; r < 8; ++r)
            out[(size_t)(mrow0 + r) * HIDDEN + ncol + s * 16] = acc[s][r];
}

// ---------------------------------------------------------------------------
extern "C" void kernel_launch(void* const* d_in, const int* in_sizes, int n_in,
                              void* d_out, int out_size, void* d_ws, size_t ws_size,
                              hipStream_t stream)
{
    const float* x  = (const float*)d_in[0];
    const float* wq = (const float*)d_in[1];
    const float* wk = (const float*)d_in[2];
    const float* wv = (const float*)d_in[3];
    const float* wo = (const float*)d_in[4];

    // Workspace layout (bf16), ~143 MB total; O aliases xb (x dead after GEMM1).
    char* p = (char*)d_ws;
    bf16* xb  = (bf16*)p; p += (size_t)TOKENS * HIDDEN * 2;
    bf16* wqb = (bf16*)p; p += (size_t)HIDDEN * HIDDEN * 2;
    bf16* wkb = (bf16*)p; p += (size_t)HIDDEN * HIDDEN * 2;
    bf16* wvb = (bf16*)p; p += (size_t)HIDDEN * HIDDEN * 2;
    bf16* wob = (bf16*)p; p += (size_t)HIDDEN * HIDDEN * 2;
    bf16* qb  = (bf16*)p; p += (size_t)TOKENS * HIDDEN * 2;
    bf16* kb  = (bf16*)p; p += (size_t)TOKENS * HIDDEN * 2;
    bf16* vb  = (bf16*)p; p += (size_t)TOKENS * HIDDEN * 2;
    bf16* ob  = xb;   // reuse

    cvt_f32_bf16<<<4096, 256, 0, stream>>>(x,  xb,  TOKENS * HIDDEN);
    cvt_f32_bf16<<<1024, 256, 0, stream>>>(wq, wqb, HIDDEN * HIDDEN);
    cvt_f32_bf16<<<1024, 256, 0, stream>>>(wk, wkb, HIDDEN * HIDDEN);
    cvt_f32_bf16<<<1024, 256, 0, stream>>>(wv, wvb, HIDDEN * HIDDEN);
    cvt_f32_bf16<<<1024, 256, 0, stream>>>(wo, wob, HIDDEN * HIDDEN);

    // 16384 wave-jobs (1024 m-tiles x 16 n-groups), 8 waves per 256-thr block.
    qkv_gemm<<<2048, 256, 0, stream>>>(xb, wqb, wkb, wvb, qb, kb, vb);

    // 1 wave per token, 4 tokens per block.
    attn_heads<<<TOKENS / 4, 128, 0, stream>>>(qb, kb, vb, ob);

    out_gemm<<<2048, 256, 0, stream>>>(ob, wob, (float*)d_out);
}
